// InstanSeg_Torchscript_37598143709702
// MI455X (gfx1250) — 8-bit matrix operand order repaired and checked by reading, compiled, not run on hardware
//
#include <hip/hip_runtime.h>
#include <stdint.h>

#define HH 512
#define WW 512
#define HWP (HH*WW)
#define KK 256
#define WSZ 32
#define SSZ 64
#define SEED_T 0.7f
#define MASK_T 0.53f
#define MIN_SIZE 10.0f
#define MEAN_T 0.0f
#define OVL_T 0.3f
#define CC_ITERS 32
#define CAP 4096

typedef __attribute__((ext_vector_type(8))) int v8i;
typedef unsigned long long u64;
typedef unsigned int u32;
typedef unsigned char u8;

#define OFF_MASK   (size_t)0
#define OFF_ROWMAX (size_t)1048576
#define OFF_MAXP   (size_t)2097152
#define OFF_LAB    (size_t)3145728
#define OFF_MW     (size_t)4194304
#define OFF_INTER  (size_t)6291456
#define OFF_CNT    (size_t)6553600
#define OFF_CVAL   (size_t)6553856
#define OFF_CIDX   (size_t)6570240
#define OFF_SCY    (size_t)6586624
#define OFF_SCX    (OFF_SCY + 1024)
#define OFF_SOY    (OFF_SCX + 1024)
#define OFF_SOX    (OFF_SOY + 1024)
#define OFF_SVALID (OFF_SOX + 1024)
#define OFF_AREA   (OFF_SVALID + 1024)
#define OFF_SUMV   (OFF_AREA + 1024)
#define OFF_REMOVE (OFF_SUMV + 1024)
#define OFF_LBL    (OFF_REMOVE + 1024)

__global__ void k_init(u32* maxp, int* lab, int* cnt) {
  int t = blockIdx.x * blockDim.x + threadIdx.x;
  if (t < HWP) { maxp[t] = 0u; lab[t] = 0; }
  if (t == 0) *cnt = 0;
}

__global__ void k_sigmoid(const float* __restrict__ x, float* __restrict__ mm) {
  int p = blockIdx.x * blockDim.x + threadIdx.x;
  if (p >= HWP) return;
  float v = x[4 * HWP + p];
  mm[p] = 1.0f / (1.0f + expf(-v));
}

__global__ void k_rowmax(const float* __restrict__ mm, float* __restrict__ rm) {
  int p = blockIdx.x * blockDim.x + threadIdx.x;
  if (p >= HWP) return;
  int y = p / WW, xq = p % WW;
  float m = -3.0e38f;
  #pragma unroll
  for (int dx = -5; dx <= 5; ++dx) {
    int xx = xq + dx;
    if (xx >= 0 && xx < WW) m = fmaxf(m, mm[y * WW + xx]);
  }
  rm[p] = m;
}

__global__ void k_peaks(const float* __restrict__ mm, const float* __restrict__ rm,
                        int* cnt, float* cval, int* cidx) {
  int p = blockIdx.x * blockDim.x + threadIdx.x;
  if (p >= HWP) return;
  int y = p / WW, xq = p % WW;
  float pool = -3.0e38f;
  #pragma unroll
  for (int dy = -5; dy <= 5; ++dy) {
    int yy = y + dy;
    if (yy >= 0 && yy < HH) pool = fmaxf(pool, rm[yy * WW + xq]);
  }
  float v = mm[p];
  if (v == pool && v > SEED_T) {
    int old = atomicAdd(cnt, 1);
    if (old < CAP) { cval[old] = v; cidx[old] = p; }
  }
}

__global__ void __launch_bounds__(1024)
k_topk(const int* cntp, const float* cval, const int* cidx,
       int* scy, int* scx, int* soy, int* sox, int* svalid) {
  __shared__ u64 keys[CAP];
  int tid = threadIdx.x;
  int cnt = *cntp; if (cnt > CAP) cnt = CAP;
  for (int t = tid; t < CAP; t += 1024) {
    u64 kkey = 0ull;
    if (t < cnt) {
      u32 fb = __float_as_uint(cval[t]);
      u32 ic = 0xFFFFFFFFu - (u32)cidx[t];
      kkey = ((u64)fb << 32) | (u64)ic;
    }
    keys[t] = kkey;
  }
  __syncthreads();
  for (int kstep = 2; kstep <= CAP; kstep <<= 1) {
    for (int jj = kstep >> 1; jj > 0; jj >>= 1) {
      for (int t = tid; t < CAP; t += 1024) {
        int ixj = t ^ jj;
        if (ixj > t) {
          u64 a = keys[t], b = keys[ixj];
          bool desc = ((t & kstep) == 0);
          if (desc ? (a < b) : (a > b)) { keys[t] = b; keys[ixj] = a; }
        }
      }
      __syncthreads();
    }
  }
  if (tid < KK) {
    u64 kkey = keys[tid];
    float v = __uint_as_float((u32)(kkey >> 32));
    int valid = (tid < cnt) && (v > SEED_T);
    int idx = valid ? (int)(0xFFFFFFFFu - (u32)kkey) : 0;
    int cy = idx / WW, cx = idx % WW;
    int wy = min(max(cy, WSZ), HH - WSZ);
    int wx = min(max(cx, WSZ), WW - WSZ);
    scy[tid] = cy; scx[tid] = cx;
    soy[tid] = wy - WSZ; sox[tid] = wx - WSZ;
    svalid[tid] = valid;
  }
}

__global__ void __launch_bounds__(256)
k_classify(const float* __restrict__ x, const float* __restrict__ wcls,
           const float* __restrict__ bcls,
           const int* scy, const int* scx, const int* soy, const int* sox,
           const int* svalid, const float* __restrict__ mm,
           u8* __restrict__ Mw2, u32* maxp, int* areas, float* sumv,
           float* __restrict__ outProbs) {
  int k = blockIdx.x;
  int tid = threadIdx.x;
  __shared__ __align__(16) u8 mtile2[8192];
  __shared__ int ared[256];
  __shared__ float sred[256];
  int valid = svalid[k];
  float w0 = wcls[0], w1 = wcls[1], w2 = wcls[2], w3 = wcls[3], bias = bcls[0];
  int oy = soy[k], ox = sox[k];
  int ox63 = ox & 63;
  float cyf = (float)scy[k], cxf = (float)scx[k];
  for (int u = tid; u < 2048; u += 256) ((u32*)mtile2)[u] = 0u;
  __syncthreads();
  int acc_a = 0; float acc_s = 0.0f;
  for (int u = tid; u < 4096; u += 256) {
    float prob = 0.0f; int mask = 0;
    int i = u >> 6, j = u & 63;
    int gy = oy + i, gx = ox + j;
    int p = gy * WW + gx;
    if (valid) {
      float d0 = x[0 * HWP + p] + (float)gy - cyf;
      float d1 = x[1 * HWP + p] + (float)gx - cxf;
      float s0 = x[2 * HWP + p];
      float s1 = x[3 * HWP + p];
      float z = w0 * d0 + w1 * d1 + w2 * s0 + w3 * s1 + bias;
      prob = 1.0f / (1.0f + expf(-z));
      mask = (prob > MASK_T) ? 1 : 0;
      if (mask) {
        atomicMax(&maxp[p], __float_as_uint(prob));
        acc_a += 1;
        acc_s += mm[p];
      }
    }
    outProbs[k * 4096 + u] = prob;
    mtile2[i * 128 + ox63 + j] = (u8)mask;
  }
  ared[tid] = acc_a; sred[tid] = acc_s;
  __syncthreads();
  for (int s2 = 128; s2 > 0; s2 >>= 1) {
    if (tid < s2) { ared[tid] += ared[tid + s2]; sred[tid] += sred[tid + s2]; }
    __syncthreads();
  }
  if (tid == 0) { areas[k] = ared[0]; sumv[k] = sred[0]; }
  for (int c = tid; c < 512; c += 256) {
    u64 gaddr = (u64)(uintptr_t)(Mw2 + (size_t)k * 8192 + (size_t)c * 16);
    u32 laddr = (u32)(uintptr_t)(mtile2 + c * 16);
    asm volatile("global_store_async_from_lds_b128 %0, %1, off"
                 :: "v"(gaddr), "v"(laddr) : "memory");
  }
  asm volatile("s_wait_asynccnt 0x0" ::: "memory");
}

__global__ void __launch_bounds__(256)
k_winners(const int* svalid, const int* soy, const int* sox,
          const u8* __restrict__ Mw2, const float* __restrict__ maxpF,
          const float* __restrict__ probs, int* lab) {
  int k = blockIdx.x;
  if (!svalid[k]) return;
  int oy = soy[k], ox = sox[k];
  int ox63 = ox & 63;
  for (int u = threadIdx.x; u < 4096; u += 256) {
    int i = u >> 6, j = u & 63;
    if (Mw2[(size_t)k * 8192 + i * 128 + ox63 + j]) {
      int p = (oy + i) * WW + (ox + j);
      float prob = probs[k * 4096 + u];
      if (prob >= maxpF[p]) atomicMax(&lab[p], k + 1);
    }
  }
}

__global__ void __launch_bounds__(32)
k_inter(const u8* __restrict__ Mw2, const int* soy, const int* sox,
        int* __restrict__ inter) {
  int lane = threadIdx.x & 31;
  int I0 = blockIdx.y * 16, J0 = blockIdx.x * 16;

  int iy0 = 1 << 30, iy1 = -(1 << 30), ix0 = 1 << 30, ix1 = -(1 << 30);
  int jy0 = 1 << 30, jy1 = -(1 << 30), jx0 = 1 << 30, jx1 = -(1 << 30);
  for (int t = 0; t < 16; ++t) {
    int a = soy[I0 + t], b = sox[I0 + t];
    iy0 = min(iy0, a); iy1 = max(iy1, a + 64);
    ix0 = min(ix0, b); ix1 = max(ix1, b + 64);
    a = soy[J0 + t]; b = sox[J0 + t];
    jy0 = min(jy0, a); jy1 = max(jy1, a + 64);
    jx0 = min(jx0, b); jx1 = max(jx1, b + 64);
  }
  int ry0 = max(iy0, jy0), ry1 = min(iy1, jy1);
  int rx0 = max(ix0, jx0), rx1 = min(ix1, jx1);

  int rA = I0 + (lane & 15);
  int abase = (lane & 16) ? 16 : 0;
  int oyA = soy[rA];
  int foxA = sox[rA] & ~63;
  const u8* baseA = Mw2 + (size_t)rA * 8192;

  int cN = J0 + (lane & 15);
  int kb = (lane & 16) ? 16 : 0;
  int oyB = soy[cN];
  int foxB = sox[cN] & ~63;
  const u8* baseB = Mw2 + (size_t)cN * 8192;

  v8i acc0 = {0, 0, 0, 0, 0, 0, 0, 0};
  v8i acc1 = {0, 0, 0, 0, 0, 0, 0, 0};

  if (ry0 < ry1 && rx0 < rx1) {
    int xs = rx0 & ~63;
    int pg = 0;
    for (int y = ry0; y < ry1; ++y) {
      int lyA = y - oyA;
      int lyB = y - oyB;
      if ((unsigned)(lyA + 1) < 64u) __builtin_prefetch(baseA + (lyA + 1) * 128, 0, 0);
      for (int xc = xs; xc < rx1; xc += 64) {
        v8i Af = {0, 0, 0, 0, 0, 0, 0, 0};
        int ca = xc - foxA;
        if ((unsigned)lyA < 64u && (unsigned)ca < 128u) {
          const u8* pA = baseA + lyA * 128 + ca + abase;
          u64 r0 = *(const u64*)(pA);
          u64 r1 = *(const u64*)(pA + 8);
          u64 r2 = *(const u64*)(pA + 32);
          u64 r3 = *(const u64*)(pA + 40);
          Af[0] = (int)(u32)r0; Af[1] = (int)(r0 >> 32);
          Af[2] = (int)(u32)r1; Af[3] = (int)(r1 >> 32);
          Af[4] = (int)(u32)r2; Af[5] = (int)(r2 >> 32);
          Af[6] = (int)(u32)r3; Af[7] = (int)(r3 >> 32);
        }
        v8i Bf = {0, 0, 0, 0, 0, 0, 0, 0};
        int cb = xc - foxB;
        if ((unsigned)lyB < 64u && (unsigned)cb < 128u) {
          const u8* pB = baseB + lyB * 128 + cb + kb;
          uint4 b0 = *(const uint4*)(pB);
          uint4 b1 = *(const uint4*)(pB + 32);
          Bf[0] = (int)b0.x; Bf[1] = (int)b0.y; Bf[2] = (int)b0.z; Bf[3] = (int)b0.w;
          Bf[4] = (int)b1.x; Bf[5] = (int)b1.y; Bf[6] = (int)b1.z; Bf[7] = (int)b1.w;
        }
        if (pg) acc1 = __builtin_amdgcn_wmma_i32_16x16x64_iu8(false, Af, false, Bf,
                                                              acc1, false, false);
        else    acc0 = __builtin_amdgcn_wmma_i32_16x16x64_iu8(false, Af, false, Bf,
                                                              acc0, false, false);
        pg ^= 1;
      }
    }
  }

  int n = J0 + (lane & 15);
  int mb = (lane & 16) ? 8 : 0;
  #pragma unroll
  for (int v2 = 0; v2 < 8; ++v2)
    inter[(size_t)(I0 + mb + v2) * KK + n] = acc0[v2] + acc1[v2];
}

__global__ void __launch_bounds__(256)
k_cc(const int* areas, const float* sumv, const int* svalid,
     const int* __restrict__ inter, int* removeF, int* lblF) {
  int i = threadIdx.x;
  __shared__ u32 adjb[KK][8];
  __shared__ int lblA[KK];
  __shared__ int lblB[KK];
  __shared__ float farea[KK];
  float ai = (float)areas[i];
  farea[i] = ai;
  __syncthreads();
  float mean = sumv[i] / fmaxf(ai, 1.0f);
  int rem = (!((mean > MEAN_T) && (ai > MIN_SIZE))) || (!svalid[i]);
  for (int w = 0; w < 8; ++w) {
    u32 bits = 0u;
    for (int t = 0; t < 32; ++t) {
      int j = w * 32 + t;
      float in = (float)inter[(size_t)i * KK + j];
      float un = ai + farea[j] - in;
      if (in / fmaxf(un, 1.0f) > OVL_T) bits |= (1u << t);
    }
    adjb[i][w] = bits;
  }
  lblA[i] = i;
  __syncthreads();
  for (int it = 0; it < CC_ITERS; ++it) {
    int m = KK;
    for (int w = 0; w < 8; ++w) {
      u32 bits = adjb[i][w];
      while (bits) {
        int t = __ffs(bits) - 1;
        bits &= bits - 1;
        int j = w * 32 + t;
        m = min(m, lblA[j]);
      }
    }
    lblB[i] = m;
    __syncthreads();
    lblA[i] = lblB[i];
    __syncthreads();
  }
  removeF[i] = rem;
  lblF[i] = lblA[i];
}

__global__ void k_final(const int* __restrict__ lab, const int* lblF,
                        const int* removeF, float* __restrict__ out) {
  int p = blockIdx.x * blockDim.x + threadIdx.x;
  if (p >= HWP) return;
  int lv = lab[p];
  int fin = 0;
  if (lv > 0) {
    int root = lblF[lv - 1] + 1;
    int rc = min(root - 1, KK - 1);
    if (!removeF[rc]) fin = root;
  }
  out[p] = (float)fin;
}

extern "C" void kernel_launch(void* const* d_in, const int* in_sizes, int n_in,
                              void* d_out, int out_size, void* d_ws, size_t ws_size,
                              hipStream_t stream) {
  (void)in_sizes; (void)n_in; (void)out_size; (void)ws_size;
  const float* x    = (const float*)d_in[0];
  const float* wcls = (const float*)d_in[1];
  const float* bcls = (const float*)d_in[2];
  float* out = (float*)d_out;
  char* ws = (char*)d_ws;

  float* mm     = (float*)(ws + OFF_MASK);
  float* rmx    = (float*)(ws + OFF_ROWMAX);
  u32*   maxp   = (u32*)  (ws + OFF_MAXP);
  int*   lab    = (int*)  (ws + OFF_LAB);
  u8*    Mw2    = (u8*)   (ws + OFF_MW);
  int*   inter  = (int*)  (ws + OFF_INTER);
  int*   cnt    = (int*)  (ws + OFF_CNT);
  float* cval   = (float*)(ws + OFF_CVAL);
  int*   cidx   = (int*)  (ws + OFF_CIDX);
  int*   scy    = (int*)  (ws + OFF_SCY);
  int*   scx    = (int*)  (ws + OFF_SCX);
  int*   soy    = (int*)  (ws + OFF_SOY);
  int*   sox    = (int*)  (ws + OFF_SOX);
  int*   svalid = (int*)  (ws + OFF_SVALID);
  int*   areas  = (int*)  (ws + OFF_AREA);
  float* sumv   = (float*)(ws + OFF_SUMV);
  int*   remF   = (int*)  (ws + OFF_REMOVE);
  int*   lblF   = (int*)  (ws + OFF_LBL);

  int blocksHW = (HWP + 255) / 256;
  k_init   <<<blocksHW, 256, 0, stream>>>(maxp, lab, cnt);
  k_sigmoid<<<blocksHW, 256, 0, stream>>>(x, mm);
  k_rowmax <<<blocksHW, 256, 0, stream>>>(mm, rmx);
  k_peaks  <<<blocksHW, 256, 0, stream>>>(mm, rmx, cnt, cval, cidx);
  k_topk   <<<1, 1024, 0, stream>>>(cnt, cval, cidx, scy, scx, soy, sox, svalid);
  k_classify<<<KK, 256, 0, stream>>>(x, wcls, bcls, scy, scx, soy, sox, svalid,
                                     mm, Mw2, maxp, areas, sumv, out + HWP);
  k_winners<<<KK, 256, 0, stream>>>(svalid, soy, sox, Mw2, (const float*)maxp,
                                    out + HWP, lab);
  dim3 gg(16, 16);
  k_inter  <<<gg, 32, 0, stream>>>(Mw2, soy, sox, inter);
  k_cc     <<<1, KK, 0, stream>>>(areas, sumv, svalid, inter, remF, lblF);
  k_final  <<<blocksHW, 256, 0, stream>>>(lab, lblF, remF, out);
}
